// MultiHeadAttention_20091857010916
// MI455X (gfx1250) — matrix operand order repaired and checked by reading, not compiled, not run on hardware
//
#include <hip/hip_runtime.h>

// ---------------------------------------------------------------------------
// Types
// ---------------------------------------------------------------------------
typedef __attribute__((ext_vector_type(16))) __bf16 v16bf;
typedef __attribute__((ext_vector_type(8)))  __bf16 v8bf;
typedef __attribute__((ext_vector_type(8)))  float  v8f;
typedef __attribute__((ext_vector_type(4)))  float  v4f;

static constexpr int DIMC = 512;
static constexpr int NHEAD = 8;
static constexpr int HDIM = 64;
static constexpr int BATCH = 4;
static constexpr int SEQ = 2048;        // l == m == 2048
static constexpr int GM = BATCH * SEQ;  // 8192 rows for the projections
static constexpr int LDT = 40;          // LDS row stride (halfs) for 32-wide K tiles (80B, 16B-aligned)

// float -> bf16, round-to-nearest-even
__device__ __forceinline__ __bf16 f2bf(float f) {
  unsigned int u = __builtin_bit_cast(unsigned int, f);
  u += 0x7FFFu + ((u >> 16) & 1u);
  unsigned short h = (unsigned short)(u >> 16);
  return __builtin_bit_cast(__bf16, h);
}

__device__ __forceinline__ v16bf cat16(v8bf lo, v8bf hi) {
  v16bf r;
#pragma unroll
  for (int i = 0; i < 8; ++i) { r[i] = lo[i]; r[8 + i] = hi[i]; }
  return r;
}

__device__ __forceinline__ v8f zero8() {
  v8f r;
#pragma unroll
  for (int i = 0; i < 8; ++i) r[i] = 0.0f;
  return r;
}

// A-fragment (16x32, M x K): lane holds row M=lane&15; element e -> K = (e<8 ? half*8+e : 16+half*8+(e-8))
// base points at the row start of a 32-wide K tile; both 8-half groups are contiguous -> two b128 loads.
__device__ __forceinline__ v16bf ldfragA(const __bf16* base, int half) {
  v8bf lo = *(const v8bf*)(base + half * 8);
  v8bf hi = *(const v8bf*)(base + 16 + half * 8);
  return cat16(lo, hi);
}

// B-fragment (32x16, K x N): lane holds col N=lane&15; element e -> K = (e<8 ? half*8+e : 16+half*8+(e-8))
// base points at the start of the 32 K-values for this N column (contiguous) -> two b128 loads.
__device__ __forceinline__ v16bf ldfragB(const __bf16* base, int half) {
  v8bf lo = *(const v8bf*)(base + half * 8);
  v8bf hi = *(const v8bf*)(base + 16 + half * 8);
  return cat16(lo, hi);
}

__device__ __forceinline__ v8f wmma_bf16(v16bf a, v16bf b, v8f c) {
  return __builtin_amdgcn_wmma_f32_16x16x32_bf16(false, a, false, b, (short)0, c, false, false);
}

// ---------------------------------------------------------------------------
// Kernel 1: projection GEMM  Y = X @ W^T  (NT), f32 in -> bf16 out
// Output written head-major: col j = d*8+n  ->  Y[((b*8+n)*SEQ + l)*64 + d]
// Grid: 64 M-tiles x 4 N-tiles = 256 blocks, 256 threads (8 waves, 4x2 wave grid)
// ---------------------------------------------------------------------------
__global__ __launch_bounds__(256)
void proj_gemm(const float* __restrict__ X, const float* __restrict__ W,
               __bf16* __restrict__ Y, float scale) {
  __shared__ __bf16 sa[128 * LDT];
  __shared__ __bf16 sb[128 * LDT];

  const int tid = threadIdx.x;
  const int wave = tid >> 5, lane = tid & 31;
  const int half = lane >> 4, lr = lane & 15;
  const int waveM = wave >> 1, waveN = wave & 1;
  const int m0 = (blockIdx.x & 63) * 128;
  const int n0 = (blockIdx.x >> 6) * 128;

  v8f acc[2][4];
#pragma unroll
  for (int mt = 0; mt < 2; ++mt)
#pragma unroll
    for (int nt = 0; nt < 4; ++nt) acc[mt][nt] = zero8();

  for (int k0 = 0; k0 < DIMC; k0 += 32) {
    __syncthreads();
    {
      const int row = tid >> 1;
      const int cb = (tid & 1) * 16;
      const v4f* px = (const v4f*)(X + (size_t)(m0 + row) * DIMC + k0 + cb);
      const v4f* pw = (const v4f*)(W + (size_t)(n0 + row) * DIMC + k0 + cb);
      v4f a0 = px[0], a1 = px[1], a2 = px[2], a3 = px[3];
      v4f b0 = pw[0], b1 = pw[1], b2 = pw[2], b3 = pw[3];
      v8bf alo, ahi, blo, bhi;
#pragma unroll
      for (int j = 0; j < 4; ++j) {
        alo[j] = f2bf(a0[j]); alo[4 + j] = f2bf(a1[j]);
        ahi[j] = f2bf(a2[j]); ahi[4 + j] = f2bf(a3[j]);
        blo[j] = f2bf(b0[j]); blo[4 + j] = f2bf(b1[j]);
        bhi[j] = f2bf(b2[j]); bhi[4 + j] = f2bf(b3[j]);
      }
      *(v8bf*)&sa[row * LDT + cb] = alo; *(v8bf*)&sa[row * LDT + cb + 8] = ahi;
      *(v8bf*)&sb[row * LDT + cb] = blo; *(v8bf*)&sb[row * LDT + cb + 8] = bhi;
    }
    __syncthreads();

    v16bf af[2];
#pragma unroll
    for (int mt = 0; mt < 2; ++mt) {
      const int row = waveM * 32 + mt * 16 + lr;
      af[mt] = ldfragA(&sa[row * LDT], half);
    }
#pragma unroll
    for (int nt = 0; nt < 4; ++nt) {
      const int ncol = waveN * 64 + nt * 16 + lr;
      v16bf bf_ = ldfragB(&sb[ncol * LDT], half);
#pragma unroll
      for (int mt = 0; mt < 2; ++mt) acc[mt][nt] = wmma_bf16(af[mt], bf_, acc[mt][nt]);
    }
  }

  // epilogue: head-major scatter with (d,n) interleave
#pragma unroll
  for (int mt = 0; mt < 2; ++mt) {
#pragma unroll
    for (int nt = 0; nt < 4; ++nt) {
      const int gn = n0 + waveN * 64 + nt * 16 + lr;
      const int nH = gn & 7, dD = gn >> 3;
#pragma unroll
      for (int v = 0; v < 8; ++v) {
        const int gm = m0 + waveM * 32 + mt * 16 + v + 8 * half;
        const int bb = gm >> 11, ll = gm & 2047;
        Y[(((size_t)(bb * NHEAD + nH) * SEQ) + ll) * HDIM + dD] = f2bf(acc[mt][nt][v] * scale);
      }
    }
  }
}

// ---------------------------------------------------------------------------
// Kernel 2: flash attention per (b, head, 128 query rows); attends over K (ref!)
// ---------------------------------------------------------------------------
__global__ __launch_bounds__(256)
void attn_kernel(const __bf16* __restrict__ QH, const __bf16* __restrict__ KH,
                 __bf16* __restrict__ XO) {
  __shared__ __bf16 kl[128 * 72];        // K chunk [m][d]   (S gemm B-frags)
  __shared__ __bf16 kt[64 * 136];        // K chunk [d][m]   (O gemm B-frags)
  __shared__ float  sbuf[8 * 16 * 132];  // per-wave raw scores
  __shared__ __bf16 pbuf[8 * 16 * 136];  // per-wave P (bf16 A-frags)
  __shared__ float  st_scale[128];
  __shared__ float  st_lsum[128];

  const int tid = threadIdx.x;
  const int wave = tid >> 5, lane = tid & 31;
  const int half = lane >> 4, lr = lane & 15;

  int blk = blockIdx.x;
  const int l0 = (blk & 15) * 128; blk >>= 4;
  const int head = blk & 7;
  const int bb = blk >> 3;

  const __bf16* Qb = QH + ((size_t)(bb * NHEAD + head) * SEQ + l0) * HDIM;
  const __bf16* Kb = KH + (size_t)(bb * NHEAD + head) * SEQ * HDIM;

  // Q fragments (wave rows wave*16..+15, d split into two 32-wide k-steps) stay in registers
  v16bf qf[2];
#pragma unroll
  for (int ks = 0; ks < 2; ++ks)
    qf[ks] = ldfragA(Qb + (size_t)(wave * 16 + lr) * HDIM + ks * 32, half);

  v8f oacc[4];
#pragma unroll
  for (int nt = 0; nt < 4; ++nt) oacc[nt] = zero8();
  float rmax = -1e30f, rsum = 0.0f;  // running stats for row (lane&15) of this wave

  for (int mc = 0; mc < SEQ; mc += 128) {
    __syncthreads();
    {  // stage K chunk into both layouts
      const int mr = tid >> 1;
      const int db = (tid & 1) * 32;
      const __bf16* kp = Kb + (size_t)(mc + mr) * HDIM + db;
#pragma unroll
      for (int i = 0; i < 4; ++i) {
        v8bf vv = *(const v8bf*)(kp + i * 8);
        *(v8bf*)&kl[mr * 72 + db + i * 8] = vv;
#pragma unroll
        for (int j = 0; j < 8; ++j) kt[(db + i * 8 + j) * 136 + mr] = vv[j];
      }
    }
    __syncthreads();

    // S = Q @ K^T for this wave's 16 rows x 128 m-cols
    v8f sacc[8];
#pragma unroll
    for (int t = 0; t < 8; ++t) sacc[t] = zero8();
#pragma unroll
    for (int t = 0; t < 8; ++t) {
      const __bf16* brow = &kl[(t * 16 + lr) * 72];
#pragma unroll
      for (int ks = 0; ks < 2; ++ks)
        sacc[t] = wmma_bf16(qf[ks], ldfragB(brow + ks * 32, half), sacc[t]);
    }

    float* sw = &sbuf[wave * 16 * 132];
#pragma unroll
    for (int t = 0; t < 8; ++t)
#pragma unroll
      for (int v = 0; v < 8; ++v) sw[(v + 8 * half) * 132 + t * 16 + lr] = sacc[t][v];
    __syncthreads();

    // online softmax: lane handles row lr, columns half*64..half*64+63
    const float* srow = &sbuf[wave * 16 * 132 + lr * 132 + half * 64];
    float cmax = -1e30f;
#pragma unroll 16
    for (int i = 0; i < 64; ++i) cmax = fmaxf(cmax, srow[i]);
    cmax = fmaxf(cmax, __shfl_xor(cmax, 16, 32));
    const float mnew = fmaxf(rmax, cmax);
    const float corr = __expf(rmax - mnew);

    __bf16* prow = &pbuf[wave * 16 * 136 + lr * 136 + half * 64];
    float csum = 0.0f;
#pragma unroll 16
    for (int i = 0; i < 64; ++i) {
      float e = __expf(srow[i] - mnew);
      csum += e;
      prow[i] = f2bf(e);
    }
    csum += __shfl_xor(csum, 16, 32);
    rsum = rsum * corr + csum;
    rmax = mnew;
    if (half == 0) st_scale[wave * 16 + lr] = corr;
    __syncthreads();

    // rescale O, then O += P @ K
    float corrv[8];
#pragma unroll
    for (int v = 0; v < 8; ++v) corrv[v] = st_scale[wave * 16 + v + 8 * half];
#pragma unroll
    for (int nt = 0; nt < 4; ++nt)
#pragma unroll
      for (int v = 0; v < 8; ++v) oacc[nt][v] *= corrv[v];

#pragma unroll
    for (int ks = 0; ks < 4; ++ks) {
      v16bf pa = ldfragA(&pbuf[wave * 16 * 136 + lr * 136 + ks * 32], half);
#pragma unroll
      for (int nt = 0; nt < 4; ++nt)
        oacc[nt] = wmma_bf16(pa, ldfragB(&kt[(nt * 16 + lr) * 136 + ks * 32], half), oacc[nt]);
    }
  }

  if (half == 0) st_lsum[wave * 16 + lr] = rsum;
  __syncthreads();
  float inv[8];
#pragma unroll
  for (int v = 0; v < 8; ++v) inv[v] = 1.0f / st_lsum[wave * 16 + v + 8 * half];

  // write x[b, l, d*8+head] as bf16
#pragma unroll
  for (int nt = 0; nt < 4; ++nt) {
    const int d = nt * 16 + lr;
    const int col = d * NHEAD + head;
#pragma unroll
    for (int v = 0; v < 8; ++v) {
      const int ll = l0 + wave * 16 + v + 8 * half;
      XO[((size_t)bb * SEQ + ll) * DIMC + col] = f2bf(oacc[nt][v] * inv[v]);
    }
  }
}

// ---------------------------------------------------------------------------
// Kernel 3: output projection  out = x @ w_o^T  (bf16 in, f32 W converted, f32 out)
// ---------------------------------------------------------------------------
__global__ __launch_bounds__(256)
void out_gemm(const __bf16* __restrict__ X, const float* __restrict__ W,
              float* __restrict__ Out) {
  __shared__ __bf16 sa[128 * LDT];
  __shared__ __bf16 sb[128 * LDT];

  const int tid = threadIdx.x;
  const int wave = tid >> 5, lane = tid & 31;
  const int half = lane >> 4, lr = lane & 15;
  const int waveM = wave >> 1, waveN = wave & 1;
  const int m0 = (blockIdx.x & 63) * 128;
  const int n0 = (blockIdx.x >> 6) * 128;

  v8f acc[2][4];
#pragma unroll
  for (int mt = 0; mt < 2; ++mt)
#pragma unroll
    for (int nt = 0; nt < 4; ++nt) acc[mt][nt] = zero8();

  for (int k0 = 0; k0 < DIMC; k0 += 32) {
    __syncthreads();
    {
      const int row = tid >> 1;
      const int cb = (tid & 1) * 16;
      const v8bf* px = (const v8bf*)(X + (size_t)(m0 + row) * DIMC + k0 + cb);
      *(v8bf*)&sa[row * LDT + cb] = px[0];
      *(v8bf*)&sa[row * LDT + cb + 8] = px[1];
      const v4f* pw = (const v4f*)(W + (size_t)(n0 + row) * DIMC + k0 + cb);
      v4f b0 = pw[0], b1 = pw[1], b2 = pw[2], b3 = pw[3];
      v8bf blo, bhi;
#pragma unroll
      for (int j = 0; j < 4; ++j) {
        blo[j] = f2bf(b0[j]); blo[4 + j] = f2bf(b1[j]);
        bhi[j] = f2bf(b2[j]); bhi[4 + j] = f2bf(b3[j]);
      }
      *(v8bf*)&sb[row * LDT + cb] = blo; *(v8bf*)&sb[row * LDT + cb + 8] = bhi;
    }
    __syncthreads();

    v16bf af[2];
#pragma unroll
    for (int mt = 0; mt < 2; ++mt) {
      const int row = waveM * 32 + mt * 16 + lr;
      af[mt] = ldfragA(&sa[row * LDT], half);
    }
#pragma unroll
    for (int nt = 0; nt < 4; ++nt) {
      const int ncol = waveN * 64 + nt * 16 + lr;
      v16bf bf_ = ldfragB(&sb[ncol * LDT], half);
#pragma unroll
      for (int mt = 0; mt < 2; ++mt) acc[mt][nt] = wmma_bf16(af[mt], bf_, acc[mt][nt]);
    }
  }

#pragma unroll
  for (int mt = 0; mt < 2; ++mt) {
#pragma unroll
    for (int nt = 0; nt < 4; ++nt) {
      const int gn = n0 + waveN * 64 + nt * 16 + lr;
#pragma unroll
      for (int v = 0; v < 8; ++v) {
        const int gm = m0 + waveM * 32 + mt * 16 + v + 8 * half;
        Out[(size_t)gm * DIMC + gn] = acc[mt][nt][v];
      }
    }
  }
}

// ---------------------------------------------------------------------------
// Host launcher
// ---------------------------------------------------------------------------
extern "C" void kernel_launch(void* const* d_in, const int* in_sizes, int n_in,
                              void* d_out, int out_size, void* d_ws, size_t ws_size,
                              hipStream_t stream) {
  const float* q  = (const float*)d_in[0];
  const float* k  = (const float*)d_in[1];
  // d_in[2] (v) and d_in[5] (w_v) are dead in the reference graph -> skipped.
  const float* wq = (const float*)d_in[3];
  const float* wk = (const float*)d_in[4];
  const float* wo = (const float*)d_in[6];
  float* out = (float*)d_out;

  const size_t headElems = (size_t)BATCH * NHEAD * SEQ * HDIM;  // 4.19M bf16 = 8MB
  __bf16* qh = (__bf16*)d_ws;
  __bf16* kh = qh + headElems;
  __bf16* xb = kh + headElems;  // 8192 x 512 bf16

  dim3 blk(256);
  // fold 1/sqrt(HEAD_DIM) = 1/8 into the Q projection
  proj_gemm<<<dim3(256), blk, 0, stream>>>(q, wq, qh, 0.125f);
  proj_gemm<<<dim3(256), blk, 0, stream>>>(k, wk, kh, 1.0f);
  attn_kernel<<<dim3(BATCH * NHEAD * (SEQ / 128)), blk, 0, stream>>>(qh, kh, xb);
  out_gemm<<<dim3(256), blk, 0, stream>>>(xb, wo, out);
}
